// EquivariantCrossAttention_27908697489440
// MI455X (gfx1250) — matrix operand order repaired and checked by reading, not compiled, not run on hardware
//
#include <hip/hip_runtime.h>
#include <math.h>

typedef __attribute__((ext_vector_type(16))) _Float16 v16h;
typedef __attribute__((ext_vector_type(8)))  float    v8f;

constexpr int Bb = 2, Nn = 2048, Ll = 128, HD = 128;
constexpr int NWAVE = 4;
constexpr float TWO_PI_F = 6.283185307179586f;
constexpr float SCALE = 0.17677669529663687f; // 1/sqrt(32)

// s_par offsets (floats)
#define P_WRQ 0
#define P_WRV 32
#define P_B1Q 64
#define P_B2Q 96
#define P_BQ  128
#define P_B1V 256
#define P_B2V 288
#define P_VFB1 320
#define P_VFG  352
#define P_VFBT 384
#define P_VFB2 416
#define P_MFB1 928
#define P_MFG  960
#define P_MFBT 992
#define P_MFB2 1024
#define P_BO   1056
#define NPAR   1088

struct EcaParams {
  const float *x, *p, *sigma, *x_h;
  const float *wr_q, *w1_q, *b1_q, *w2_q, *b2_q;
  const float *wr_v, *w1_v, *b1_v, *w2_v, *b2_v;
  const float *wq, *bq;
  const float *cf_w1, *cf_b1, *cf_g, *cf_bt, *cf_w2, *cf_b2;
  const float *vf_w1, *vf_b1, *vf_g, *vf_bt, *vf_w2, *vf_b2;
  const float *mf_w1, *mf_b1, *mf_g, *mf_bt, *mf_w2, *mf_b2;
  const float *wo, *bo;
  const float *kws, *vws;
  float *out;
};

__device__ __forceinline__ v8f vzero() {
  v8f z = {0.f,0.f,0.f,0.f,0.f,0.f,0.f,0.f};
  return z;
}

__device__ __forceinline__ v8f wmma16(v16h a, v16h b, v8f c) {
  return __builtin_amdgcn_wmma_f32_16x16x32_f16(false, a, false, b, (short)0, c, false, false);
}

// gelu (tanh approx) via hw exp + fast rcp: tanh(u) = 1 - 2/(exp(2u)+1)
__device__ __forceinline__ float gelu1(float x) {
  float u2 = 1.5957691216057308f * (x + 0.044715f * x * x * x); // 2*0.7978845608*(...)
  float e = __expf(u2);
  float t = 1.0f - 2.0f * __builtin_amdgcn_rcpf(e + 1.0f);
  return 0.5f * x * (1.0f + t);
}

__device__ __forceinline__ void gelu8(v8f& d) {
#pragma unroll
  for (int i = 0; i < 8; ++i) d[i] = gelu1(d[i]);
}

__device__ __forceinline__ void addb(v8f& d, const float* b, int lane, int cb) {
  float bv = b[cb + (lane & 15)];
#pragma unroll
  for (int i = 0; i < 8; ++i) d[i] += bv;
}

__device__ __forceinline__ float rsum16(float t) {
  t += __shfl_xor(t, 1);
  t += __shfl_xor(t, 2);
  t += __shfl_xor(t, 4);
  t += __shfl_xor(t, 8);
  return t;
}

// LayerNorm over 32 columns held in two D-fragments (per-row stats via 16-lane butterfly)
__device__ __forceinline__ void layernorm32(v8f& d0, v8f& d1, const float* g, const float* bt, int lane) {
  int c0 = lane & 15, c1 = 16 + (lane & 15);
  float g0 = g[c0], g1 = g[c1], t0 = bt[c0], t1 = bt[c1];
#pragma unroll
  for (int i = 0; i < 8; ++i) {
    float s1 = rsum16(d0[i] + d1[i]);
    float s2 = rsum16(d0[i]*d0[i] + d1[i]*d1[i]);
    float mu = s1 * (1.0f/32.0f);
    float var = s2 * (1.0f/32.0f) - mu*mu;
    float rstd = rsqrtf(var + 1e-6f);
    d0[i] = (d0[i]-mu)*rstd*g0 + t0;
    d1[i] = (d1[i]-mu)*rstd*g1 + t1;
  }
}

// store two D-frags (16x32 f32) into per-wave LDS tile as f16
__device__ __forceinline__ void xstore(_Float16* tp, int lane, const v8f& d, int colbase) {
  int col = colbase + (lane & 15);
  int r0 = (lane >> 4) * 8;
#pragma unroll
  for (int i = 0; i < 8; ++i) tp[(r0 + i) * 32 + col] = (_Float16)d[i];
}

// load an A-fragment (16x32 f16, ISA layout) from the per-wave tile
__device__ __forceinline__ v16h xload(const _Float16* tp, int lane) {
  int M = lane & 15;
  int base = (lane >> 4) * 8;
  v16h a;
#pragma unroll
  for (int j = 0; j < 8; ++j) {
    a[j]     = tp[M*32 + base + j];
    a[j + 8] = tp[M*32 + 16 + base + j];
  }
  return a;
}

__device__ __forceinline__ v16h loadfrag(const _Float16* s_frag, int f, int lane) {
  return *(const v16h*)(s_frag + (f << 9) + (lane << 4));
}

// ---------------- precompute k = a@wk+bk, v = a@wv+bv ----------------
__global__ __launch_bounds__(128) void eca_kv_kernel(const float* __restrict__ a,
    const float* __restrict__ wk, const float* __restrict__ bk,
    const float* __restrict__ wv, const float* __restrict__ bv,
    float* __restrict__ kout, float* __restrict__ vout) {
  int bl = blockIdx.x;      // 0..B*L-1
  int c  = threadIdx.x;     // 0..127
  __shared__ float arow[128];
  arow[c] = a[bl * 128 + c];
  __syncthreads();
  float sk = bk[c], sv = bv[c];
#pragma unroll 4
  for (int k = 0; k < 128; ++k) {
    float av = arow[k];
    sk += av * wk[k * 128 + c];
    sv += av * wv[k * 128 + c];
  }
  kout[bl * 128 + c] = sk;
  vout[bl * 128 + c] = sv;
}

// ---------------- main fused kernel ----------------
__global__ __launch_bounds__(128) void eca_main_kernel(EcaParams P) {
  __shared__ __align__(32) _Float16 s_frag[38 * 512];   // 38 B-fragments, pre-swizzled
  __shared__ float s_par[NPAR];
  __shared__ __align__(16) _Float16 s_tile[NWAVE][512]; // per-wave 16x32 transpose tile
  __shared__ float s_gb[1024];                          // FiLM conditioner tile (16x64)
  __shared__ float s_yt[2048];                          // softmax-weighted v accumulator (16x128)
  __shared__ float s_S[64];                             // per (row,h) softmax denom

  const int tid  = threadIdx.x;
  const int lane = tid & 31;
  const int w    = tid >> 5;
  const int wg   = blockIdx.x;
  const int bb   = wg / (Nn / 16);
  const int n0   = (wg % (Nn / 16)) * 16;
  const int lc   = lane & 15;          // column within 16-wide tile
  const int base8 = (lane >> 4) * 8;   // row base for D-frags / K base for A-frags

  // ---- copy params to LDS ----
  {
    struct { int off; const float* src; int n; } cp[16] = {
      {P_WRQ, P.wr_q, 32}, {P_WRV, P.wr_v, 32},
      {P_B1Q, P.b1_q, 32}, {P_B2Q, P.b2_q, 32}, {P_BQ, P.bq, 128},
      {P_B1V, P.b1_v, 32}, {P_B2V, P.b2_v, 32},
      {P_VFB1, P.vf_b1, 32}, {P_VFG, P.vf_g, 32}, {P_VFBT, P.vf_bt, 32},
      {P_VFB2, P.vf_b2, 512},
      {P_MFB1, P.mf_b1, 32}, {P_MFG, P.mf_g, 32}, {P_MFBT, P.mf_bt, 32},
      {P_MFB2, P.mf_b2, 32}, {P_BO, P.bo, 32},
    };
    for (int e = 0; e < 16; ++e)
      for (int i = tid; i < cp[e].n; i += 128)
        s_par[cp[e].off + i] = cp[e].src[i];
  }

  // ---- pre-swizzle weight B-fragments into LDS (wave handles frags w, w+4, ...) ----
  for (int f = w; f < 38; f += NWAVE) {
    const float* Wp; int ncols, t;
    if (f < 2)       { Wp = P.w1_q;  ncols = 32;  t = f; }
    else if (f < 4)  { Wp = P.w2_q;  ncols = 32;  t = f - 2; }
    else if (f < 12) { Wp = P.wq;    ncols = 128; t = f - 4; }
    else if (f < 14) { Wp = P.w1_v;  ncols = 32;  t = f - 12; }
    else if (f < 16) { Wp = P.w2_v;  ncols = 32;  t = f - 14; }
    else if (f < 18) { Wp = P.vf_w1; ncols = 32;  t = f - 16; }
    else if (f < 34) { Wp = P.vf_w2; ncols = 256; t = f - 18; }
    else if (f < 36) { Wp = P.mf_w1; ncols = 32;  t = f - 34; }
    else             { Wp = P.mf_w2; ncols = 32;  t = f - 36; }
    int Ncol = 16 * t + lc;
    int kb = (lane >> 4) << 4;
#pragma unroll
    for (int j = 0; j < 16; ++j)
      s_frag[f * 512 + lane * 16 + j] = (_Float16)Wp[(base8 + (j & 7) + ((j & 8) << 1)) * ncols + Ncol];
  }

  // ---- FiLM conditioner gb = FFN(x_h): h = gelu(x@w1+b1); LN; @w2+b2 (temps in s_yt) ----
  for (int o = tid; o < 512; o += 128) {   // h tile (16x32) -> s_yt[0..511]
    int row = o >> 5, c = o & 31;
    const float* xr = P.x_h + (size_t)(bb * Nn + n0 + row) * 32;
    float acc = P.cf_b1[c];
#pragma unroll 4
    for (int k = 0; k < 32; ++k) acc += xr[k] * P.cf_w1[k * 32 + c];
    s_yt[o] = gelu1(acc);
  }
  __syncthreads();
  if (tid < 16) {                          // LN stats -> s_yt[512..575]
    float s1 = 0.f, s2 = 0.f;
    for (int k = 0; k < 32; ++k) { float v = s_yt[tid * 32 + k]; s1 += v; s2 += v * v; }
    float mu = s1 * (1.0f/32.0f);
    float var = s2 * (1.0f/32.0f) - mu * mu;
    s_yt[512 + tid] = mu;
    s_yt[544 + tid] = rsqrtf(var + 1e-6f);
  }
  __syncthreads();
  for (int o = tid; o < 1024; o += 128) {  // gb tile (16x64) -> s_gb
    int row = o >> 6, c = o & 63;
    float mu = s_yt[512 + row], rstd = s_yt[544 + row];
    float acc = P.cf_b2[c];
#pragma unroll 4
    for (int k = 0; k < 32; ++k) {
      float hn = (s_yt[row * 32 + k] - mu) * rstd * P.cf_g[k] + P.cf_bt[k];
      acc += hn * P.cf_w2[k * 64 + c];
    }
    s_gb[o] = acc;
  }
  __syncthreads();
  // zero accumulators
  for (int o = tid; o < 2048; o += 128) s_yt[o] = 0.f;
  if (tid < 64) s_S[tid] = 0.f;
  __syncthreads();

  // ---- per-wave loop over latents (no-max softmax: logits are small) ----
  _Float16* tp = &s_tile[w][0];
  const int n = n0 + lc;  // this lane's row (both halves share the row for A-frags)
  const float xv0 = P.x[(size_t)(bb * Nn + n) * 2 + 0];
  const float xv1 = P.x[(size_t)(bb * Nn + n) * 2 + 1];

  for (int l = w; l < Ll; l += NWAVE) {
    float p0 = P.p[(size_t)(bb * Ll + l) * 2 + 0];
    float p1 = P.p[(size_t)(bb * Ll + l) * 2 + 1];
    float inv0 = xv0 - p0, inv1 = xv1 - p1;
    float sg = P.sigma[bb * Ll + l];
    float gw = -(inv0 * inv0 + inv1 * inv1) / (2.0f * sg * sg);

    // RFF features directly in A-fragment layout
    v16h aq, av;
#pragma unroll
    for (int j = 0; j < 8; ++j) {
      int t = base8 + j;
      float pq = TWO_PI_F * (inv0 * s_par[P_WRQ + t] + inv1 * s_par[P_WRQ + 16 + t]);
      aq[j] = (_Float16)__sinf(pq); aq[j + 8] = (_Float16)__cosf(pq);
      float pv = TWO_PI_F * (inv0 * s_par[P_WRV + t] + inv1 * s_par[P_WRV + 16 + t]);
      av[j] = (_Float16)__sinf(pv); av[j + 8] = (_Float16)__cosf(pv);
    }

    // ===== q path: e_q = (gelu(feat@w1+b1))@w2+b2 =====
    v8f d0 = wmma16(aq, loadfrag(s_frag, 0, lane), vzero());
    v8f d1 = wmma16(aq, loadfrag(s_frag, 1, lane), vzero());
    addb(d0, s_par + P_B1Q, lane, 0); addb(d1, s_par + P_B1Q, lane, 16);
    gelu8(d0); gelu8(d1);
    xstore(tp, lane, d0, 0); xstore(tp, lane, d1, 16);
    v16h ah = xload(tp, lane);
    d0 = wmma16(ah, loadfrag(s_frag, 2, lane), vzero());
    d1 = wmma16(ah, loadfrag(s_frag, 3, lane), vzero());
    addb(d0, s_par + P_B2Q, lane, 0); addb(d1, s_par + P_B2Q, lane, 16);
    xstore(tp, lane, d0, 0); xstore(tp, lane, d1, 16);
    v16h aeq = xload(tp, lane);

    // ===== v path: e_v embedding, FiLM, FFN hidden =====
    d0 = wmma16(av, loadfrag(s_frag, 12, lane), vzero());
    d1 = wmma16(av, loadfrag(s_frag, 13, lane), vzero());
    addb(d0, s_par + P_B1V, lane, 0); addb(d1, s_par + P_B1V, lane, 16);
    gelu8(d0); gelu8(d1);
    xstore(tp, lane, d0, 0); xstore(tp, lane, d1, 16);
    v16h ahv = xload(tp, lane);
    d0 = wmma16(ahv, loadfrag(s_frag, 14, lane), vzero());
    d1 = wmma16(ahv, loadfrag(s_frag, 15, lane), vzero());
    addb(d0, s_par + P_B2V, lane, 0); addb(d1, s_par + P_B2V, lane, 16);
    // FiLM with gb (g in cols 0..31, b in cols 32..63)
#pragma unroll
    for (int i = 0; i < 8; ++i) {
      int row = base8 + i;
      d0[i] = d0[i] * (1.0f + s_gb[row * 64 + lc])      + s_gb[row * 64 + 32 + lc];
      d1[i] = d1[i] * (1.0f + s_gb[row * 64 + 16 + lc]) + s_gb[row * 64 + 48 + lc];
    }
    xstore(tp, lane, d0, 0); xstore(tp, lane, d1, 16);
    v16h aev = xload(tp, lane);
    d0 = wmma16(aev, loadfrag(s_frag, 16, lane), vzero());
    d1 = wmma16(aev, loadfrag(s_frag, 17, lane), vzero());
    addb(d0, s_par + P_VFB1, lane, 0); addb(d1, s_par + P_VFB1, lane, 16);
    gelu8(d0); gelu8(d1);
    layernorm32(d0, d1, s_par + P_VFG, s_par + P_VFBT, lane);
    xstore(tp, lane, d0, 0); xstore(tp, lane, d1, 16);
    v16h ahf = xload(tp, lane);

    // ===== per-head: q proj + logit, vgb on demand, v FiLM, mixer FFN, LDS accumulate =====
#pragma unroll 1
    for (int h = 0; h < 4; ++h) {
      // q projection for this head + attention logit
      v8f q0 = wmma16(aeq, loadfrag(s_frag, 4 + 2 * h, lane), vzero());
      v8f q1 = wmma16(aeq, loadfrag(s_frag, 5 + 2 * h, lane), vzero());
      addb(q0, s_par + P_BQ, lane, 32 * h); addb(q1, s_par + P_BQ, lane, 32 * h + 16);
      float k0 = P.kws[(size_t)(bb * Ll + l) * HD + 32 * h + lc];
      float k1 = P.kws[(size_t)(bb * Ll + l) * HD + 32 * h + 16 + lc];
      float pe8[8];
#pragma unroll
      for (int i = 0; i < 8; ++i)
        pe8[i] = __expf(rsum16(q0[i] * k0 + q1[i] * k1) * SCALE + gw);

      // vgb for this head's columns
      v8f vg0 = wmma16(ahf, loadfrag(s_frag, 18 + 2 * h,     lane), vzero());
      v8f vg1 = wmma16(ahf, loadfrag(s_frag, 18 + 2 * h + 1, lane), vzero());
      v8f vb0 = wmma16(ahf, loadfrag(s_frag, 26 + 2 * h,     lane), vzero());
      v8f vb1 = wmma16(ahf, loadfrag(s_frag, 26 + 2 * h + 1, lane), vzero());
      addb(vg0, s_par + P_VFB2, lane, 32 * h);
      addb(vg1, s_par + P_VFB2, lane, 32 * h + 16);
      addb(vb0, s_par + P_VFB2, lane, 128 + 32 * h);
      addb(vb1, s_par + P_VFB2, lane, 128 + 32 * h + 16);
      float vv0 = P.vws[(size_t)(bb * Ll + l) * HD + 32 * h + lc];
      float vv1 = P.vws[(size_t)(bb * Ll + l) * HD + 32 * h + 16 + lc];
      v8f vn0, vn1;
#pragma unroll
      for (int i = 0; i < 8; ++i) {
        vn0[i] = vv0 * (1.0f + vg0[i]) + vb0[i];
        vn1[i] = vv1 * (1.0f + vg1[i]) + vb1[i];
      }
      // mixer FFN on this head's 16x32 slice
      xstore(tp, lane, vn0, 0); xstore(tp, lane, vn1, 16);
      v16h am = xload(tp, lane);
      v8f m0 = wmma16(am, loadfrag(s_frag, 34, lane), vzero());
      v8f m1 = wmma16(am, loadfrag(s_frag, 35, lane), vzero());
      addb(m0, s_par + P_MFB1, lane, 0); addb(m1, s_par + P_MFB1, lane, 16);
      gelu8(m0); gelu8(m1);
      layernorm32(m0, m1, s_par + P_MFG, s_par + P_MFBT, lane);
      xstore(tp, lane, m0, 0); xstore(tp, lane, m1, 16);
      v16h ahm = xload(tp, lane);
      v8f f0 = wmma16(ahm, loadfrag(s_frag, 36, lane), vzero());
      v8f f1 = wmma16(ahm, loadfrag(s_frag, 37, lane), vzero());
      addb(f0, s_par + P_MFB2, lane, 0); addb(f1, s_par + P_MFB2, lane, 16);

      // softmax-weighted accumulation into LDS (ds_add_f32 atomics)
      if (lc == 0) {
#pragma unroll
        for (int i = 0; i < 8; ++i)
          atomicAdd(&s_S[(base8 + i) * 4 + h], pe8[i]);
      }
#pragma unroll
      for (int i = 0; i < 8; ++i) {
        int row = base8 + i;
        atomicAdd(&s_yt[row * 128 + (2 * h) * 16 + lc],     pe8[i] * f0[i]);
        atomicAdd(&s_yt[row * 128 + (2 * h + 1) * 16 + lc], pe8[i] * f1[i]);
      }
    }
  }

  __syncthreads();

  // ---- normalize ----
  for (int o = tid; o < 2048; o += 128) {
    int row = o >> 7, col = o & 127;
    s_yt[o] = s_yt[o] * __builtin_amdgcn_rcpf(s_S[row * 4 + (col >> 5)]);
  }
  __syncthreads();

  // ---- output projection y(16x128) @ wo(128x32) + bo, wave 0 only (wo frags from global) ----
  if (w == 0) {
    v8f o0 = vzero(), o1 = vzero();
    int M = lane & 15;
    int kb = (lane >> 4) << 4;
#pragma unroll
    for (int c = 0; c < 4; ++c) {
      v16h ay, bw0, bw1;
#pragma unroll
      for (int j = 0; j < 8; ++j) {
        ay[j]     = (_Float16)s_yt[M * 128 + 32 * c + base8 + j];
        ay[j + 8] = (_Float16)s_yt[M * 128 + 32 * c + 16 + base8 + j];
      }
#pragma unroll
      for (int j = 0; j < 16; ++j) {
        int kk = 32 * c + base8 + (j & 7) + ((j & 8) << 1);
        bw0[j] = (_Float16)P.wo[kk * 32 + lc];
        bw1[j] = (_Float16)P.wo[kk * 32 + 16 + lc];
      }
      o0 = wmma16(ay, bw0, o0);
      o1 = wmma16(ay, bw1, o1);
    }
    float bo0 = s_par[P_BO + lc], bo1 = s_par[P_BO + 16 + lc];
#pragma unroll
    for (int i = 0; i < 8; ++i) {
      int nrow = n0 + base8 + i;
      P.out[(size_t)(bb * Nn + nrow) * 32 + lc]      = o0[i] + bo0;
      P.out[(size_t)(bb * Nn + nrow) * 32 + 16 + lc] = o1[i] + bo1;
    }
  }
}

extern "C" void kernel_launch(void* const* d_in, const int* in_sizes, int n_in,
                              void* d_out, int out_size, void* d_ws, size_t ws_size,
                              hipStream_t stream) {
  (void)in_sizes; (void)n_in; (void)out_size; (void)ws_size;
  const float* x     = (const float*)d_in[0];
  const float* p     = (const float*)d_in[1];
  const float* a     = (const float*)d_in[2];
  const float* sigma = (const float*)d_in[3];
  const float* x_h   = (const float*)d_in[4];
  const float* wr_q  = (const float*)d_in[5];
  const float* w1_q  = (const float*)d_in[6];
  const float* b1_q  = (const float*)d_in[7];
  const float* w2_q  = (const float*)d_in[8];
  const float* b2_q  = (const float*)d_in[9];
  const float* wr_v  = (const float*)d_in[10];
  const float* w1_v  = (const float*)d_in[11];
  const float* b1_v  = (const float*)d_in[12];
  const float* w2_v  = (const float*)d_in[13];
  const float* b2_v  = (const float*)d_in[14];
  const float* wq    = (const float*)d_in[15];
  const float* bq    = (const float*)d_in[16];
  const float* wk    = (const float*)d_in[17];
  const float* bk    = (const float*)d_in[18];
  const float* wv    = (const float*)d_in[19];
  const float* bv    = (const float*)d_in[20];
  const float* cf_w1 = (const float*)d_in[21];
  const float* cf_b1 = (const float*)d_in[22];
  const float* cf_g  = (const float*)d_in[23];
  const float* cf_bt = (const float*)d_in[24];
  const float* cf_w2 = (const float*)d_in[25];
  const float* cf_b2 = (const float*)d_in[26];
  const float* vf_w1 = (const float*)d_in[27];
  const float* vf_b1 = (const float*)d_in[28];
  const float* vf_g  = (const float*)d_in[29];
  const float* vf_bt = (const float*)d_in[30];
  const float* vf_w2 = (const float*)d_in[31];
  const float* vf_b2 = (const float*)d_in[32];
  const float* mf_w1 = (const float*)d_in[33];
  const float* mf_b1 = (const float*)d_in[34];
  const float* mf_g  = (const float*)d_in[35];
  const float* mf_bt = (const float*)d_in[36];
  const float* mf_w2 = (const float*)d_in[37];
  const float* mf_b2 = (const float*)d_in[38];
  const float* wo    = (const float*)d_in[39];
  const float* bo    = (const float*)d_in[40];

  float* kws = (float*)d_ws;
  float* vws = kws + (size_t)Bb * Ll * HD;

  eca_kv_kernel<<<Bb * Ll, 128, 0, stream>>>(a, wk, bk, wv, bv, kws, vws);

  EcaParams P;
  P.x = x; P.p = p; P.sigma = sigma; P.x_h = x_h;
  P.wr_q = wr_q; P.w1_q = w1_q; P.b1_q = b1_q; P.w2_q = w2_q; P.b2_q = b2_q;
  P.wr_v = wr_v; P.w1_v = w1_v; P.b1_v = b1_v; P.w2_v = w2_v; P.b2_v = b2_v;
  P.wq = wq; P.bq = bq;
  P.cf_w1 = cf_w1; P.cf_b1 = cf_b1; P.cf_g = cf_g; P.cf_bt = cf_bt; P.cf_w2 = cf_w2; P.cf_b2 = cf_b2;
  P.vf_w1 = vf_w1; P.vf_b1 = vf_b1; P.vf_g = vf_g; P.vf_bt = vf_bt; P.vf_w2 = vf_w2; P.vf_b2 = vf_b2;
  P.mf_w1 = mf_w1; P.mf_b1 = mf_b1; P.mf_g = mf_g; P.mf_bt = mf_bt; P.mf_w2 = mf_w2; P.mf_b2 = mf_b2;
  P.wo = wo; P.bo = bo;
  P.kws = kws; P.vws = vws;
  P.out = (float*)d_out;

  eca_main_kernel<<<Bb * (Nn / 16), 128, 0, stream>>>(P);
}
